// Encoder_46033459478804
// MI455X (gfx1250) — matrix operand order repaired and checked by reading, not compiled, not run on hardware
//
#include <hip/hip_runtime.h>
#include <cstdint>
#include <cstddef>

// ---------------------------------------------------------------------------
// MI455X / gfx1250, wave32. f16 WMMA (16x16x32, fp32 accum) for all GEMMs.
// Fused attention keeps the 64xS fp32 score strip in LDS (256KB of 320KB WGP
// LDS). Tile movement uses GLOBAL_LOAD_ASYNC_TO_LDS_B128 (ASYNCcnt) with
// double buffering.
// ---------------------------------------------------------------------------

typedef __attribute__((ext_vector_type(16))) _Float16 v16h;
typedef __attribute__((ext_vector_type(8)))  float    v8f;
typedef __attribute__((ext_vector_type(4)))  int      v4i;
typedef _Float16 h16;

#define DEVINL __device__ __forceinline__

#if defined(__HIP_DEVICE_COMPILE__) && \
    __has_builtin(__builtin_amdgcn_global_load_async_to_lds_b128)
#define HAVE_ASYNC_COPY 1
#else
#define HAVE_ASYNC_COPY 0
#endif

// Copy 32 bytes (16 halves) global -> LDS for this lane.
// Builtin params are int4* in the HIP cuda_device / cuda_shared address
// spaces; generic pointers convert implicitly in HIP. INST_OFFSET is added to
// both the global and the LDS address (ISA 08 section 4.4), so the second
// 16B transfer reuses the same pointers with offset 16.
DEVINL void cp32B(h16* __restrict__ dst, const h16* __restrict__ src) {
#if HAVE_ASYNC_COPY
    __builtin_amdgcn_global_load_async_to_lds_b128((v4i*)src, (v4i*)dst, 0, 0);
    __builtin_amdgcn_global_load_async_to_lds_b128((v4i*)src, (v4i*)dst, 16, 0);
#else
#pragma unroll
    for (int i = 0; i < 16; ++i) dst[i] = src[i];
#endif
}

DEVINL void async_wait0() {
#if HAVE_ASYNC_COPY
#if __has_builtin(__builtin_amdgcn_s_wait_asynccnt)
    __builtin_amdgcn_s_wait_asynccnt(0);
#else
    asm volatile("s_wait_asynccnt 0x0" ::: "memory");
#endif
#endif
}

DEVINL float wave_max(float v) {
#pragma unroll
    for (int m = 16; m > 0; m >>= 1) v = fmaxf(v, __shfl_xor(v, m, 32));
    return v;
}
DEVINL float wave_sum(float v) {
#pragma unroll
    for (int m = 16; m > 0; m >>= 1) v += __shfl_xor(v, m, 32);
    return v;
}

// A fragment: 16x32 f16 tile at (row0,k0) of row-major [rows x ld].
// ISA 7.12.2: lanes 0-15 -> M; halves 0..7 = K(kb..), 8..15 = K(16+kb..),
// kb = 0 (lanes 0-15) or 8 (lanes 16-31).
DEVINL v16h frag_a(const h16* base, int ld, int row0, int k0, int lane) {
    const h16* p = base + (size_t)(row0 + (lane & 15)) * ld + k0 + ((lane >> 4) << 3);
    v16h f;
#pragma unroll
    for (int i = 0; i < 8; ++i) f[i] = p[i];
#pragma unroll
    for (int i = 0; i < 8; ++i) f[8 + i] = p[16 + i];
    return f;
}

// B fragment (NT): B[k][n] = W[n][k], W row-major [N x K] (k contiguous).
// lane -> column N; halves 0..7 = K(kb..), 8..15 = K(16+kb..),
// kb = 0 (lanes 0-15) or 8 (lanes 16-31), matching frag_a.
DEVINL v16h frag_b_nt(const h16* base, int ld, int col0, int k0, int lane) {
    const h16* p = base + (size_t)(col0 + (lane & 15)) * ld + k0 + ((lane >> 4) << 3);
    v16h f;
#pragma unroll
    for (int i = 0; i < 16; ++i) f[i] = p[i < 8 ? i : i + 8];
    return f;
}

// B fragment (NN): source row-major [K x N] (e.g. V tile), column per lane.
DEVINL v16h frag_b_nn(const h16* base, int ld, int k0, int col0, int lane) {
    const h16* p = base + (size_t)(k0 + ((lane >> 4) << 3)) * ld + col0 + (lane & 15);
    v16h f;
#pragma unroll
    for (int i = 0; i < 16; ++i) f[i] = p[(i < 8 ? i : i + 8) * ld];
    return f;
}

// A fragment from fp32 LDS (entmax probabilities), converted to f16 on the fly.
DEVINL v16h frag_a_f32(const float* base, int ld, int row0, int k0, int lane) {
    const float* p = base + (size_t)(row0 + (lane & 15)) * ld + k0 + ((lane >> 4) << 3);
    v16h f;
#pragma unroll
    for (int i = 0; i < 8; ++i) f[i] = (h16)p[i];
#pragma unroll
    for (int i = 0; i < 8; ++i) f[8 + i] = (h16)p[16 + i];
    return f;
}

DEVINL v8f wmma16(v16h a, v16h b, v8f c) {
    return __builtin_amdgcn_wmma_f32_16x16x32_f16(false, a, false, b, (short)0, c,
                                                  false, false);
}

// ---------------------------------------------------------------------------
// fp32 -> f16 cast (weights, once per launch)
// ---------------------------------------------------------------------------
__global__ __launch_bounds__(256)
void cast_f32_to_f16(const float* __restrict__ in, h16* __restrict__ out, int n) {
    int i = blockIdx.x * 256 + threadIdx.x;
    if (i < n) out[i] = (h16)in[i];
}

// ---------------------------------------------------------------------------
// LayerNorm over D=512, one row per workgroup (256 threads, 2 elems/thread)
// ---------------------------------------------------------------------------
__global__ __launch_bounds__(256)
void ln512(const float* __restrict__ x, const float* __restrict__ g,
           const float* __restrict__ bta, void* __restrict__ out, int f32out) {
    const int D = 512;
    const int row = blockIdx.x, tid = threadIdx.x, lane = tid & 31, w = tid >> 5;
    __shared__ float red[8];
    const float* xr = x + (size_t)row * D;
    float v0 = xr[tid], v1 = xr[tid + 256];
    float s = wave_sum(v0 + v1);
    if (lane == 0) red[w] = s;
    __syncthreads();
    float mu = 0.f;
#pragma unroll
    for (int i = 0; i < 8; ++i) mu += red[i];
    mu *= (1.f / 512.f);
    float d0 = v0 - mu, d1 = v1 - mu;
    __syncthreads();
    float vs = wave_sum(d0 * d0 + d1 * d1);
    if (lane == 0) red[w] = vs;
    __syncthreads();
    float var = 0.f;
#pragma unroll
    for (int i = 0; i < 8; ++i) var += red[i];
    var *= (1.f / 512.f);
    float rstd = rsqrtf(var + 1e-5f);
    float o0 = d0 * rstd * g[tid] + bta[tid];
    float o1 = d1 * rstd * g[tid + 256] + bta[tid + 256];
    if (f32out) {
        ((float*)out)[(size_t)row * D + tid]       = o0;
        ((float*)out)[(size_t)row * D + tid + 256] = o1;
    } else {
        ((h16*)out)[(size_t)row * D + tid]       = (h16)o0;
        ((h16*)out)[(size_t)row * D + tid + 256] = (h16)o1;
    }
}

// ---------------------------------------------------------------------------
// Tiled NT GEMM: C[m,n] = sum_k A[m,k] * W[n,k] (+bias). 64x64x64 tiles,
// 256 threads = 8 waves, 2 WMMA C-tiles per wave. Double-buffered LDS with
// async global->LDS copies: [wait][barrier][issue t+1][wmma t].
// EPI: 0 = f16 out, 1 = Mish -> f16 out, 2 = fp32 out with residual add.
// ---------------------------------------------------------------------------
constexpr int BK = 64, LDT = BK + 8;

template <int EPI>
__global__ __launch_bounds__(256)
void gemm_nt(const h16* __restrict__ A, const h16* __restrict__ W,
             const float* __restrict__ bias, const float* __restrict__ res,
             void* __restrict__ out, int M, int N, int K) {
    __shared__ h16 As[2][64 * LDT];
    __shared__ h16 Ws[2][64 * LDT];
    const int m0 = blockIdx.y * 64, n0 = blockIdx.x * 64;
    const int tid = threadIdx.x, lane = tid & 31, w = tid >> 5;
    const int tn = (w & 3) * 16;          // C-tile column
    const int tm = (w >> 2) * 16;         // C-tile rows: tm and tm+32
    const int lrow = tid >> 2, loff = (tid & 3) << 4;
    v8f z8 = {0.f, 0.f, 0.f, 0.f, 0.f, 0.f, 0.f, 0.f};
    v8f acc[2] = {z8, z8};

    auto issue = [&](int buf, int k0) {
        cp32B(&As[buf][lrow * LDT + loff], A + (size_t)(m0 + lrow) * K + k0 + loff);
        cp32B(&Ws[buf][lrow * LDT + loff], W + (size_t)(n0 + lrow) * K + k0 + loff);
    };

    const int T = K / BK;
    issue(0, 0);
    for (int t = 0; t < T; ++t) {
        const int cur = t & 1;
        async_wait0();
        __syncthreads();                   // tile t ready; tile t-1 reads done
        if (t + 1 < T) issue(cur ^ 1, (t + 1) * BK);
#pragma unroll
        for (int kk = 0; kk < BK; kk += 32) {
            v16h bf = frag_b_nt(Ws[cur], LDT, tn, kk, lane);
            v16h a0 = frag_a(As[cur], LDT, tm, kk, lane);
            v16h a1 = frag_a(As[cur], LDT, tm + 32, kk, lane);
            acc[0] = wmma16(a0, bf, acc[0]);
            acc[1] = wmma16(a1, bf, acc[1]);
        }
    }

    const int nl = n0 + tn + (lane & 15);
    const float bn = bias[nl];
#pragma unroll
    for (int t = 0; t < 2; ++t) {
#pragma unroll
        for (int r = 0; r < 8; ++r) {
            int ml = m0 + tm + t * 32 + ((lane >> 4) << 3) + r;
            float v = acc[t][r] + bn;
            size_t idx = (size_t)ml * N + nl;
            if (EPI == 0) {
                ((h16*)out)[idx] = (h16)v;
            } else if (EPI == 1) {            // Mish(v) = v * tanh(softplus(v))
                float sp = (v > 20.f) ? v : log1pf(expf(v));
                ((h16*)out)[idx] = (h16)(v * tanhf(sp));
            } else {
                ((float*)out)[idx] = res[idx] + v;
            }
        }
    }
}

// ---------------------------------------------------------------------------
// Fused entmax-1.5 attention. One workgroup owns 64 query rows of one
// (batch, head). Full 64 x S fp32 score strip lives in LDS; QK^T and att*V
// both via WMMA with double-buffered async K/V staging; entmax tau solved by
// 26-iteration bisection per row.
// Dynamic LDS: sc[64*S] f32 | Qs[64*72] | KV[2][64*72]  (~283KB)
// NOTE: the KV double buffer is selected by an *integer offset* from a single
// LDS base pointer so clang can prove addrspace(3) and emit ds_load (a
// pointer-array select here previously lowered the reads to flat_load).
// ---------------------------------------------------------------------------
__global__ __launch_bounds__(256)
void attn_entmax15(const h16* __restrict__ q16, const h16* __restrict__ k16,
                   const h16* __restrict__ vv16, h16* __restrict__ o16,
                   int S, int D) {
    constexpr int LDA = 72;
    constexpr int KVSZ = 64 * LDA;
    extern __shared__ char smem[];
    float* sc = (float*)smem;                          // [64][S]
    h16* Qs   = (h16*)(smem + (size_t)64 * S * 4);     // [64][LDA]
    h16* KVb  = Qs + KVSZ;                             // [2][64][LDA]

    const int q0 = blockIdx.x * 64, h = blockIdx.y, b = blockIdx.z;
    const int tid = threadIdx.x, lane = tid & 31, w = tid >> 5;
    const int hoff = h * 64;
    const int lrow = tid >> 2, loff = (tid & 3) << 4;
    const int tn = (w & 3) * 16, tm = (w >> 2) * 16;
    const int NT = S / 64;

    auto issueK = [&](int buf, int ct) {
        cp32B(KVb + buf * KVSZ + lrow * LDA + loff,
              k16 + (size_t)(b * S + ct * 64 + lrow) * D + hoff + loff);
    };
    auto issueV = [&](int buf, int kt) {
        cp32B(KVb + buf * KVSZ + lrow * LDA + loff,
              vv16 + (size_t)(b * S + kt * 64 + lrow) * D + hoff + loff);
    };

    // prologue: stage Q block + first K tile (async)
    cp32B(&Qs[lrow * LDA + loff],
          q16 + (size_t)(b * S + q0 + lrow) * D + hoff + loff);
    issueK(0, 0);

    // ---- phase 1: scores = (Q K^T) / sqrt(64), into LDS strip ----
    for (int ct = 0; ct < NT; ++ct) {
        const int cur = ct & 1;
        async_wait0();
        __syncthreads();
        if (ct + 1 < NT) issueK(cur ^ 1, ct + 1);
        const h16* kv = KVb + cur * KVSZ;
        v8f z8 = {0.f, 0.f, 0.f, 0.f, 0.f, 0.f, 0.f, 0.f};
        v8f acc0 = z8, acc1 = z8;
#pragma unroll
        for (int kk = 0; kk < 64; kk += 32) {
            v16h bf = frag_b_nt(kv, LDA, tn, kk, lane);   // B[d][k] = K[k][d]
            v16h a0 = frag_a(Qs, LDA, tm, kk, lane);
            v16h a1 = frag_a(Qs, LDA, tm + 32, kk, lane);
            acc0 = wmma16(a0, bf, acc0);
            acc1 = wmma16(a1, bf, acc1);
        }
#pragma unroll
        for (int r = 0; r < 8; ++r) {
            int ml = tm + ((lane >> 4) << 3) + r;
            int nl = ct * 64 + tn + (lane & 15);
            sc[(size_t)ml * S + nl]        = acc0[r] * 0.125f;
            sc[(size_t)(ml + 32) * S + nl] = acc1[r] * 0.125f;
        }
    }
    __syncthreads();

    // ---- phase 2: entmax-1.5 per row (one wave per row, 8 rows/wave) ----
    for (int rr = 0; rr < 8; ++rr) {
        int row = w * 8 + rr;
        float* Z = sc + (size_t)row * S;
        float zl[32];
        float zm = -3.0e38f;
#pragma unroll
        for (int j = 0; j < 32; ++j) { zl[j] = Z[lane + 32 * j]; zm = fmaxf(zm, zl[j]); }
        zm = wave_max(zm);
#pragma unroll
        for (int j = 0; j < 32; ++j) zl[j] = (zl[j] - zm) * 0.5f;  // s_i, max = 0
        // f(tau) = sum relu(s_i - tau)^2 - 1, root in [-1, 0]
        float lo = -1.f, hi = 0.f;
        for (int it = 0; it < 26; ++it) {
            float tau = 0.5f * (lo + hi);
            float s = 0.f;
#pragma unroll
            for (int j = 0; j < 32; ++j) {
                float t = zl[j] - tau; t = t > 0.f ? t : 0.f; s += t * t;
            }
            s = wave_sum(s);
            if (s >= 1.f) lo = tau; else hi = tau;
        }
        float tau = 0.5f * (lo + hi);
        float s = 0.f;
#pragma unroll
        for (int j = 0; j < 32; ++j) {
            float t = zl[j] - tau; t = t > 0.f ? t : 0.f; zl[j] = t * t; s += zl[j];
        }
        s = wave_sum(s);
        float inv = 1.f / s;  // snap sum to exactly 1
#pragma unroll
        for (int j = 0; j < 32; ++j) Z[lane + 32 * j] = zl[j] * inv;
    }

    // ---- phase 3: out = att @ V (A from fp32 LDS, cvt f16 on the fly) ----
    issueV(0, 0);  // KV buffers are free; sc handoff guarded by loop barrier
    v8f z8 = {0.f, 0.f, 0.f, 0.f, 0.f, 0.f, 0.f, 0.f};
    v8f acc0 = z8, acc1 = z8;
    for (int kt = 0; kt < NT; ++kt) {
        const int cur = kt & 1;
        async_wait0();
        __syncthreads();
        if (kt + 1 < NT) issueV(cur ^ 1, kt + 1);
        const h16* kv = KVb + cur * KVSZ;
#pragma unroll
        for (int kk = 0; kk < 64; kk += 32) {
            v16h bf = frag_b_nn(kv, LDA, kk, tn, lane);   // B[k][d] = V[k][d]
            v16h a0 = frag_a_f32(sc, S, tm, kt * 64 + kk, lane);
            v16h a1 = frag_a_f32(sc, S, tm + 32, kt * 64 + kk, lane);
            acc0 = wmma16(a0, bf, acc0);
            acc1 = wmma16(a1, bf, acc1);
        }
    }
#pragma unroll
    for (int r = 0; r < 8; ++r) {
        int ml = tm + ((lane >> 4) << 3) + r;
        int nl = tn + (lane & 15);
        o16[(size_t)(b * S + q0 + ml) * D + hoff + nl]      = (h16)acc0[r];
        o16[(size_t)(b * S + q0 + ml + 32) * D + hoff + nl] = (h16)acc1[r];
    }
}

// ---------------------------------------------------------------------------
// Host-side orchestration (all launches on `stream`, graph-capture safe)
// ---------------------------------------------------------------------------
extern "C" void kernel_launch(void* const* d_in, const int* in_sizes, int n_in,
                              void* d_out, int out_size, void* d_ws, size_t ws_size,
                              hipStream_t stream) {
    (void)in_sizes; (void)n_in; (void)out_size; (void)ws_size;
    const int B = 8, S = 1024, D = 512, F = 2048;
    const int MS = B * S;  // 8192 rows

    const float* x    = (const float*)d_in[0];
    const float* Wq   = (const float*)d_in[1];
    const float* bq   = (const float*)d_in[2];
    const float* Wk   = (const float*)d_in[3];
    const float* bk   = (const float*)d_in[4];
    const float* Wv   = (const float*)d_in[5];
    const float* bv   = (const float*)d_in[6];
    const float* Wo   = (const float*)d_in[7];
    const float* bo   = (const float*)d_in[8];
    const float* ln1g = (const float*)d_in[9];
    const float* ln1b = (const float*)d_in[10];
    const float* W1   = (const float*)d_in[11];
    const float* bf1  = (const float*)d_in[12];
    const float* W2   = (const float*)d_in[13];
    const float* bf2  = (const float*)d_in[14];
    const float* ln2g = (const float*)d_in[15];
    const float* ln2b = (const float*)d_in[16];
    const float* lnfg = (const float*)d_in[17];
    const float* lnfb = (const float*)d_in[18];

    char* ws = (char*)d_ws;
    size_t off = 0;
    auto take = [&](size_t bytes) -> char* {
        char* p = ws + off;
        off = (off + bytes + 255) & ~(size_t)255;
        return p;
    };
    h16* wq16 = (h16*)take((size_t)D * D * 2);
    h16* wk16 = (h16*)take((size_t)D * D * 2);
    h16* wv16 = (h16*)take((size_t)D * D * 2);
    h16* wo16 = (h16*)take((size_t)D * D * 2);
    h16* w116 = (h16*)take((size_t)F * D * 2);
    h16* w216 = (h16*)take((size_t)D * F * 2);
    h16* y16  = (h16*)take((size_t)MS * D * 2);   // LN1 out
    h16* q16  = (h16*)take((size_t)MS * D * 2);
    h16* k16  = (h16*)take((size_t)MS * D * 2);
    h16* v16  = (h16*)take((size_t)MS * D * 2);
    h16* o16  = (h16*)take((size_t)MS * D * 2);   // attention out
    float* x1 = (float*)take((size_t)MS * D * 4); // after attn residual
    h16* y216 = (h16*)take((size_t)MS * D * 2);   // LN2 out
    h16* ff16 = (h16*)take((size_t)MS * F * 2);   // Mish(FFN1) out
    float* x2 = (float*)take((size_t)MS * D * 4); // after FFN residual

    auto castN = [&](const float* src, h16* dst, int n) {
        cast_f32_to_f16<<<dim3((n + 255) / 256), dim3(256), 0, stream>>>(src, dst, n);
    };
    castN(Wq, wq16, D * D);
    castN(Wk, wk16, D * D);
    castN(Wv, wv16, D * D);
    castN(Wo, wo16, D * D);
    castN(W1, w116, F * D);
    castN(W2, w216, D * F);

    // LN1 -> f16
    ln512<<<dim3(MS), dim3(256), 0, stream>>>(x, ln1g, ln1b, y16, 0);

    // Q/K/V projections
    dim3 gDD(D / 64, MS / 64);
    gemm_nt<0><<<gDD, dim3(256), 0, stream>>>(y16, wq16, bq, nullptr, q16, MS, D, D);
    gemm_nt<0><<<gDD, dim3(256), 0, stream>>>(y16, wk16, bk, nullptr, k16, MS, D, D);
    gemm_nt<0><<<gDD, dim3(256), 0, stream>>>(y16, wv16, bv, nullptr, v16, MS, D, D);

    // fused entmax-1.5 attention (~283KB dynamic LDS per workgroup)
    size_t shm = (size_t)64 * S * 4 + 3 * (size_t)64 * 72 * 2;
    attn_entmax15<<<dim3(S / 64, 8, B), dim3(256), shm, stream>>>(q16, k16, v16, o16, S, D);

    // O projection + residual (fp32)
    gemm_nt<2><<<gDD, dim3(256), 0, stream>>>(o16, wo16, bo, x, x1, MS, D, D);

    // LN2 -> f16
    ln512<<<dim3(MS), dim3(256), 0, stream>>>(x1, ln2g, ln2b, y216, 0);

    // FFN1 (+Mish) and FFN2 (+residual)
    dim3 gFD(F / 64, MS / 64);
    gemm_nt<1><<<gFD, dim3(256), 0, stream>>>(y216, w116, bf1, nullptr, ff16, MS, F, D);
    gemm_nt<2><<<gDD, dim3(256), 0, stream>>>(ff16, w216, bf2, x1, x2, MS, D, F);

    // final LayerNorm -> fp32 d_out
    ln512<<<dim3(MS), dim3(256), 0, stream>>>(x2, lnfg, lnfb, d_out, 1);
}
